// SubgraphCompressorDecompressor_1717986918822
// MI455X (gfx1250) — matrix operand order repaired and checked by reading, not compiled, not run on hardware
//
#include <hip/hip_runtime.h>
#include <hip/hip_bf16.h>

#define F 128

typedef __attribute__((ext_vector_type(16))) _Float16 v16h;
typedef __attribute__((ext_vector_type(8)))  float    v8f;

union AFrag { v16h v; _Float16 h[16]; };
union BFrag { v16h v; uint4 q[2]; };

static __device__ __forceinline__ unsigned pack2h(float a, float b) {
    union { _Float16 h[2]; unsigned u; } r;
    r.h[0] = (_Float16)a; r.h[1] = (_Float16)b;
    return r.u;
}

// ---------------------------------------------------------------------------
// Utility: zero a float buffer (count divisible by 4)
// ---------------------------------------------------------------------------
__global__ void zero_f32(float* __restrict__ p, int n4) {
    int i = blockIdx.x * blockDim.x + threadIdx.x;
    if (i < n4) {
        float4 z; z.x = 0.f; z.y = 0.f; z.z = 0.f; z.w = 0.f;
        ((float4*)p)[i] = z;
    }
}

// ---------------------------------------------------------------------------
// Degree: deg[dst[e]] += 1
// ---------------------------------------------------------------------------
__global__ void deg_accum(const int* __restrict__ dst, float* __restrict__ deg, int E) {
    int e = blockIdx.x * blockDim.x + threadIdx.x;
    if (e < E) atomicAdd(&deg[dst[e]], 1.0f);
}

// ---------------------------------------------------------------------------
// Scatter-add: one wave (32 lanes) per edge, float4 per lane (32*4 = 128)
// ---------------------------------------------------------------------------
__global__ __launch_bounds__(256) void scatter_add(
    const float* __restrict__ X, const int* __restrict__ src,
    const int* __restrict__ dst, float* __restrict__ Agg, int E)
{
    int e = blockIdx.x * 8 + (threadIdx.x >> 5);
    if (e >= E) return;
    int lane = threadIdx.x & 31;
    int s = src[e], d = dst[e];
    if (e + 64 < E)  // stream-ahead prefetch of a future source row
        __builtin_prefetch(X + (size_t)src[e + 64] * F + lane * 4, 0, 1);
    float4 v = *(const float4*)(X + (size_t)s * F + lane * 4);
    float* o = Agg + (size_t)d * F + lane * 4;
    atomicAdd(o + 0, v.x);
    atomicAdd(o + 1, v.y);
    atomicAdd(o + 2, v.z);
    atomicAdd(o + 3, v.w);
}

// ---------------------------------------------------------------------------
// Pack one f32 [128][128] weight (row-major W[k][n]) into the per-lane WMMA
// B-fragment layout for V_WMMA_F32_16X16X32_F16:
//   tile(kt,nt): 32x16 block; lanes 0-15 hold K=k0+{0..7,16..23} for col n0+lane,
//                lanes 16-31 hold K=k0+{8..15,24..31} for col n0+lane-16,
//   8 packed-dword VGPRs per lane, contiguous in memory (2x b128 loads).
// ---------------------------------------------------------------------------
__global__ void pack_weight(const float* __restrict__ W, unsigned* __restrict__ P) {
    int t = blockIdx.x * blockDim.x + threadIdx.x;   // 1024 lanes total
    if (t >= 1024) return;
    int lane = t & 31, nt = (t >> 5) & 7, kt = t >> 8;
    int col = nt * 16 + (lane & 15);
    int kb  = kt * 32 + ((lane < 16) ? 0 : 8);
    unsigned* o = P + ((size_t)(kt * 8 + nt) * 32 + lane) * 8;
#pragma unroll
    for (int v = 0; v < 8; ++v)
        o[v] = pack2h(W[(size_t)(kb + 2 * v + (v & 4) * 2) * F + col],
                      W[(size_t)(kb + 2 * v + (v & 4) * 2 + 1) * F + col]);
}

// ---------------------------------------------------------------------------
// Fused SAGE dense stage:
//   Out[i,:] = relu?( X[i,:]@Wself + (Agg[i,:]*rcp(max(deg,1)))@Wneigh + bias )
// Both packed weight matrices staged in LDS once per block (8 waves share);
// one wave computes a 16x128 row tile via 64 v_wmma_f32_16x16x32_f16 fed by
// ds_load_b128 B-fragments.
// ---------------------------------------------------------------------------
__global__ __launch_bounds__(256) void sage_gemm(
    const float* __restrict__ X, const float* __restrict__ Agg,
    const float* __restrict__ Deg,
    const unsigned* __restrict__ Wsp, const unsigned* __restrict__ Wnp,
    const float* __restrict__ bias, float* __restrict__ Out,
    int n, int do_relu)
{
    __shared__ uint4 lds_w[4096];  // 64 KB: [0..2047] Wself, [2048..4095] Wneigh

    // cooperative stage of both packed matrices into LDS
    {
        const uint4* gs = (const uint4*)Wsp;
        const uint4* gn = (const uint4*)Wnp;
#pragma unroll
        for (int i = 0; i < 8; ++i) {
            int idx = threadIdx.x + i * 256;
            lds_w[idx]        = gs[idx];
            lds_w[idx + 2048] = gn[idx];
        }
    }
    __syncthreads();

    const int lane = threadIdx.x & 31;
    const int wave = threadIdx.x >> 5;
    const int tile = blockIdx.x * 8 + wave;
    if (tile * 16 >= n) return;
    const int m0 = tile * 16;
    int row = m0 + (lane & 15);
    if (row >= n) row = n - 1;                 // clamp loads for ragged tail
    const int off = (lane < 16) ? 0 : 8;       // A-frag K sub-offset per half-wave

    const float* xr = X   + (size_t)row * F;
    const float* ar = Agg + (size_t)row * F;
    // fast reciprocal (v_rcp_f32) -- no IEEE divide expansion needed here
    const float inv = __builtin_amdgcn_rcpf(fmaxf(Deg[row], 1.0f));

    // A fragments (16x32 f16): lane<16 -> K {0..7,16..23}+k0; lane>=16 -> +8
    AFrag ax[4], am[4];
#pragma unroll
    for (int kt = 0; kt < 4; ++kt) {
        const int k0 = kt * 32 + off;
        float fx[16], fa[16];
        ((float4*)fx)[0] = *(const float4*)(xr + k0);
        ((float4*)fx)[1] = *(const float4*)(xr + k0 + 4);
        ((float4*)fx)[2] = *(const float4*)(xr + k0 + 16);
        ((float4*)fx)[3] = *(const float4*)(xr + k0 + 20);
        ((float4*)fa)[0] = *(const float4*)(ar + k0);
        ((float4*)fa)[1] = *(const float4*)(ar + k0 + 4);
        ((float4*)fa)[2] = *(const float4*)(ar + k0 + 16);
        ((float4*)fa)[3] = *(const float4*)(ar + k0 + 20);
#pragma unroll
        for (int j = 0; j < 16; ++j) {
            ax[kt].h[j] = (_Float16)fx[j];
            am[kt].h[j] = (_Float16)(fa[j] * inv);
        }
    }

#pragma unroll
    for (int nt = 0; nt < 8; ++nt) {
        const float bv = bias[nt * 16 + (lane & 15)];
        v8f acc;
#pragma unroll
        for (int r = 0; r < 8; ++r) acc[r] = bv;
#pragma unroll
        for (int kt = 0; kt < 4; ++kt) {
            const int fi = ((kt * 8 + nt) * 32 + lane) * 2;  // uint4 index
            BFrag bs, bn;
            bs.q[0] = lds_w[fi];
            bs.q[1] = lds_w[fi + 1];
            bn.q[0] = lds_w[fi + 2048];
            bn.q[1] = lds_w[fi + 2049];
            acc = __builtin_amdgcn_wmma_f32_16x16x32_f16(
                false, ax[kt].v, false, bs.v, (short)0, acc, false, false);
            acc = __builtin_amdgcn_wmma_f32_16x16x32_f16(
                false, am[kt].v, false, bn.v, (short)0, acc, false, false);
        }
        const int col   = nt * 16 + (lane & 15);
        const int mbase = m0 + ((lane < 16) ? 0 : 8);
#pragma unroll
        for (int r = 0; r < 8; ++r) {
            float val = acc[r];
            if (do_relu) val = fmaxf(val, 0.0f);
            int m = mbase + r;
            if (m < n) Out[(size_t)m * F + col] = val;
        }
    }
}

// ---------------------------------------------------------------------------
// score = sigmoid(H[i,:].w_score + b_score); key = monotone uint transform
// ---------------------------------------------------------------------------
__global__ __launch_bounds__(256) void score_keys(
    const float* __restrict__ H, const float* __restrict__ wsc,
    const float* __restrict__ bsc, unsigned* __restrict__ keys, int n)
{
    int i = blockIdx.x * 8 + (threadIdx.x >> 5);
    if (i >= n) return;
    int lane = threadIdx.x & 31;
    float4 h = *(const float4*)(H + (size_t)i * F + lane * 4);
    float4 w = *(const float4*)(wsc + lane * 4);
    float s = h.x * w.x + h.y * w.y + h.z * w.z + h.w * w.w;
#pragma unroll
    for (int o = 16; o; o >>= 1) s += __shfl_xor(s, o, 32);
    if (lane == 0) {
        float sc = 1.0f / (1.0f + __expf(-(s + bsc[0])));
        union { float f; unsigned u; } c; c.f = sc;
        unsigned b = c.u;
        keys[i] = b ^ ((b >> 31) ? 0xFFFFFFFFu : 0x80000000u);
    }
}

// ---------------------------------------------------------------------------
// Exact radix top-k select: state = {prefixKey, remaining, tieCnt, quota}
// ---------------------------------------------------------------------------
__global__ void init_state(unsigned* __restrict__ st, unsigned k) {
    st[0] = 0u; st[1] = k; st[2] = 0u; st[3] = 0u;
}

__global__ void zero_hist(unsigned* __restrict__ hist) { hist[threadIdx.x] = 0u; }

__global__ void hist_pass(const unsigned* __restrict__ keys,
                          const unsigned* __restrict__ st,
                          unsigned* __restrict__ hist, int n, int level)
{
    int i = blockIdx.x * blockDim.x + threadIdx.x;
    if (i >= n) return;
    unsigned prefix   = st[0];
    unsigned maskHigh = (level == 3) ? 0u : (0xFFFFFFFFu << (8 * (level + 1)));
    unsigned key = keys[i];
    if (((key ^ prefix) & maskHigh) == 0u) {
        unsigned bkt = (key >> (8 * level)) & 0xFFu;
        atomicAdd(&hist[bkt], 1u);
    }
}

__global__ void select_step(unsigned* __restrict__ st,
                            const unsigned* __restrict__ hist, int level)
{
    if (threadIdx.x != 0) return;
    unsigned remaining = st[1];
    unsigned cum = 0; int chosen = 0;
    for (int b = 255; b >= 0; --b) {
        unsigned c = hist[b];
        if (cum + c >= remaining) { chosen = b; remaining -= cum; break; }
        cum += c;
    }
    st[0] |= ((unsigned)chosen) << (8 * level);
    st[1]  = remaining;
    if (level == 0) { st[3] = remaining; st[2] = 0u; }  // quota for ties
}

__global__ void flag_select(const unsigned* __restrict__ keys,
                            unsigned* __restrict__ st,
                            unsigned* __restrict__ flags, int n)
{
    int i = blockIdx.x * blockDim.x + threadIdx.x;
    if (i >= n) return;
    unsigned T = st[0], quota = st[3];
    unsigned key = keys[i];
    unsigned f = 0u;
    if (key > T) f = 1u;
    else if (key == T) { if (atomicAdd(&st[2], 1u) < quota) f = 1u; }
    flags[i] = f;
}

__global__ void build_newval(const float* __restrict__ H,
                             const unsigned* __restrict__ flags,
                             float* __restrict__ NV, int n)
{
    int idx = blockIdx.x * blockDim.x + threadIdx.x;  // one per float4
    int i = idx >> 5; int c = (idx & 31) * 4;
    if (i >= n) return;
    float4 v;
    if (flags[i]) v = *(const float4*)(H + (size_t)i * F + c);
    else { v.x = 0.f; v.y = 0.f; v.z = 0.f; v.w = 0.f; }
    *(float4*)(NV + (size_t)i * F + c) = v;
}

// ---------------------------------------------------------------------------
// Host-side orchestration
// ---------------------------------------------------------------------------
extern "C" void kernel_launch(void* const* d_in, const int* in_sizes, int n_in,
                              void* d_out, int out_size, void* d_ws, size_t ws_size,
                              hipStream_t stream)
{
    const float* x   = (const float*)d_in[0];
    const int*   src = (const int*)d_in[1];
    const int*   dst = (const int*)d_in[2];
    const float* ws1 = (const float*)d_in[3];
    const float* wn1 = (const float*)d_in[4];
    const float* b1  = (const float*)d_in[5];
    const float* ws2 = (const float*)d_in[6];
    const float* wn2 = (const float*)d_in[7];
    const float* b2  = (const float*)d_in[8];
    const float* wsc = (const float*)d_in[9];
    const float* bsc = (const float*)d_in[10];
    const float* wu1 = (const float*)d_in[11];
    const float* wnu1= (const float*)d_in[12];
    const float* bu1 = (const float*)d_in[13];
    const float* wu2 = (const float*)d_in[14];
    const float* wnu2= (const float*)d_in[15];
    const float* bu2 = (const float*)d_in[16];
    float* out = (float*)d_out;

    const int N = in_sizes[0] / F;
    const int E = in_sizes[1];
    const unsigned K = (unsigned)((N / 4 > 1) ? (N / 4) : 1);
    const size_t NF = (size_t)N * F;

    // carve workspace
    char* wp = (char*)d_ws;
    auto carve = [&](size_t bytes) -> void* {
        void* p = (void*)wp;
        wp += (bytes + 255) & ~(size_t)255;
        return p;
    };
    float*    agg   = (float*)carve(NF * 4);
    float*    h1    = (float*)carve(NF * 4);
    float*    h2    = (float*)carve(NF * 4);
    float*    nv    = (float*)carve(NF * 4);
    float*    d1    = (float*)carve(NF * 4);
    float*    deg   = (float*)carve((size_t)N * 4);
    unsigned* keys  = (unsigned*)carve((size_t)N * 4);
    unsigned* flags = (unsigned*)carve((size_t)N * 4);
    unsigned* hist  = (unsigned*)carve(256 * 4);
    unsigned* state = (unsigned*)carve(4 * 4);
    unsigned* pk    = (unsigned*)carve(8 * 8192 * 4);  // 8 packed f16 weights
    (void)ws_size; (void)n_in; (void)out_size;

    unsigned* pws1  = pk + 0 * 8192;
    unsigned* pwn1  = pk + 1 * 8192;
    unsigned* pws2  = pk + 2 * 8192;
    unsigned* pwn2  = pk + 3 * 8192;
    unsigned* pwu1  = pk + 4 * 8192;
    unsigned* pwnu1 = pk + 5 * 8192;
    unsigned* pwu2  = pk + 6 * 8192;
    unsigned* pwnu2 = pk + 7 * 8192;

    // pack weights to WMMA B-fragment layout (f16)
    pack_weight<<<4, 256, 0, stream>>>(ws1,  pws1);
    pack_weight<<<4, 256, 0, stream>>>(wn1,  pwn1);
    pack_weight<<<4, 256, 0, stream>>>(ws2,  pws2);
    pack_weight<<<4, 256, 0, stream>>>(wn2,  pwn2);
    pack_weight<<<4, 256, 0, stream>>>(wu1,  pwu1);
    pack_weight<<<4, 256, 0, stream>>>(wnu1, pwnu1);
    pack_weight<<<4, 256, 0, stream>>>(wu2,  pwu2);
    pack_weight<<<4, 256, 0, stream>>>(wnu2, pwnu2);

    const int zb_nf  = (int)((NF / 4 + 255) / 256);
    const int zb_n   = (N / 4 + 255) / 256;
    const int eb     = (E + 7) / 8;
    const int ebl    = (E + 255) / 256;
    const int tiles  = (N + 15) / 16;
    const int gb     = (tiles + 7) / 8;
    const int rb     = (N + 7) / 8;        // one wave per row
    const int nb     = (N + 255) / 256;

    // degree (reused by all 4 layers)
    zero_f32<<<zb_n, 256, 0, stream>>>(deg, N / 4);
    deg_accum<<<ebl, 256, 0, stream>>>(dst, deg, E);

    // ---- compress layer 1: x -> h1 (relu) ----
    zero_f32<<<zb_nf, 256, 0, stream>>>(agg, (int)(NF / 4));
    scatter_add<<<eb, 256, 0, stream>>>(x, src, dst, agg, E);
    sage_gemm<<<gb, 256, 0, stream>>>(x, agg, deg, pws1, pwn1, b1, h1, N, 1);

    // ---- compress layer 2: h1 -> h2 ----
    zero_f32<<<zb_nf, 256, 0, stream>>>(agg, (int)(NF / 4));
    scatter_add<<<eb, 256, 0, stream>>>(h1, src, dst, agg, E);
    sage_gemm<<<gb, 256, 0, stream>>>(h1, agg, deg, pws2, pwn2, b2, h2, N, 0);

    // ---- score + exact radix top-k -> gate mask ----
    score_keys<<<rb, 256, 0, stream>>>(h2, wsc, bsc, keys, N);
    init_state<<<1, 1, 0, stream>>>(state, K);
    for (int level = 3; level >= 0; --level) {
        zero_hist<<<1, 256, 0, stream>>>(hist);
        hist_pass<<<nb, 256, 0, stream>>>(keys, state, hist, N, level);
        select_step<<<1, 1, 0, stream>>>(state, hist, level);
    }
    flag_select<<<nb, 256, 0, stream>>>(keys, state, flags, N);
    build_newval<<<rb * 8, 256, 0, stream>>>(h2, flags, nv, N);

    // ---- decompress layer 1: nv -> d1 (relu) ----
    zero_f32<<<zb_nf, 256, 0, stream>>>(agg, (int)(NF / 4));
    scatter_add<<<eb, 256, 0, stream>>>(nv, src, dst, agg, E);
    sage_gemm<<<gb, 256, 0, stream>>>(nv, agg, deg, pwu1, pwnu1, bu1, d1, N, 1);

    // ---- decompress layer 2: d1 -> out ----
    zero_f32<<<zb_nf, 256, 0, stream>>>(agg, (int)(NF / 4));
    scatter_add<<<eb, 256, 0, stream>>>(d1, src, dst, agg, E);
    sage_gemm<<<gb, 256, 0, stream>>>(d1, agg, deg, pwu2, pwnu2, bu2, out, N, 0);
}
